// CorrKernel_29798483099793
// MI455X (gfx1250) — matrix operand order repaired and checked by reading, not compiled, not run on hardware
//
#include <hip/hip_runtime.h>

// CDNA5 / gfx1250: wave32, WMMA 16x16x32 bf16, f32 accumulate.
typedef __attribute__((ext_vector_type(16))) __bf16 v16bf;
typedef __attribute__((ext_vector_type(8)))  float  v8f;

namespace {
constexpr int  kB    = 4;
constexpr int  kC    = 256;
constexpr int  kH    = 96;
constexpr int  kW    = 160;
constexpr int  kMaxD = 4;
constexpr int  kP    = 9;            // patch side
constexpr int  kPP   = kP * kP;      // 81 output channels
constexpr int  kXT   = kW / 16;      // 10 exact x-tiles
constexpr long kPlane = (long)kH * kW;  // channel stride (elements)
}  // namespace

// One wave per (b, y, xtile, dy). Computes the 9-diagonal band of
// G[x, j] = sum_c f0[c, y, x] * f1[c, y+dy, j] via two 16x16 WMMA tiles
// (j in [x0-4, x0+11] and [x0+12, x0+27]), bf16x3 split for ~fp32 accuracy.
__global__ __launch_bounds__(32)
void corr_wmma_bf16x3(const float* __restrict__ f0,
                      const float* __restrict__ f1,
                      float* __restrict__ out) {
  int id = blockIdx.x;
  const int dyi = id % kP;  id /= kP;   // 0..8  -> dy = dyi-4
  const int xt  = id % kXT; id /= kXT;  // 0..9
  const int y   = id % kH;  id /= kH;   // 0..95
  const int b   = id;                   // 0..3

  const int lane = threadIdx.x;   // 0..31
  const int half = lane >> 4;     // lane group (0 or 1)
  const int lm   = lane & 15;     // m (A rows) / n (B cols)

  const int  x0  = xt * 16;
  const int  y1  = y + dyi - kMaxD;
  const bool yok = (unsigned)y1 < (unsigned)kH;
  const int  y1c = yok ? y1 : 0;

  // A fragment source: column x0+lm of f0[b, :, y, :]
  const float* a_col = f0 + ((long)b * kC * kH + y) * kW + (x0 + lm);
  // B fragment source: row y1 of f1[b, :, :, :]
  const float* b_row = f1 + ((long)b * kC * kH + y1c) * kW;

  const int  j0  = x0 - 4 + lm;    // N-tile 0 column
  const int  j1  = x0 + 12 + lm;   // N-tile 1 column
  const bool ok0 = yok && (unsigned)j0 < (unsigned)kW;
  const bool ok1 = yok && (unsigned)j1 < (unsigned)kW;

  v8f acc0 = {};
  v8f acc1 = {};

  for (int kk = 0; kk < kC; kk += 32) {
    if (kk + 32 < kC) {  // pull next K-chunk toward L2/L0 (speculative)
      __builtin_prefetch(a_col + (long)(kk + 32 + 8 * half) * kPlane, 0, 1);
      __builtin_prefetch(b_row + (long)(kk + 32 + 16 * half) * kPlane + (x0 + lm), 0, 1);
    }

    v16bf Ah, Al, B0h, B0l, B1h, B1l;
#pragma unroll
    for (int e = 0; e < 16; ++e) {
      // A (16x32 bf16): lane<16 holds K = {0..7,16..23}, lane>=16 K = {8..15,24..31}
      const int   kA = kk + 8 * half + (e < 8 ? e : e + 8);
      const float av = a_col[(long)kA * kPlane];
      const __bf16 ah = (__bf16)av;
      Ah[e] = ah;
      Al[e] = (__bf16)(av - (float)ah);

      // B (32x16 bf16): lane<16 holds K = {0..7,16..23}, lane>=16 K = {8..15,24..31}
      const int   kBk = kk + 8 * half + (e < 8 ? e : e + 8);
      const float b0v = ok0 ? b_row[(long)kBk * kPlane + j0] : 0.0f;
      const float b1v = ok1 ? b_row[(long)kBk * kPlane + j1] : 0.0f;
      const __bf16 b0h = (__bf16)b0v;
      const __bf16 b1h = (__bf16)b1v;
      B0h[e] = b0h;
      B0l[e] = (__bf16)(b0v - (float)b0h);
      B1h[e] = b1h;
      B1l[e] = (__bf16)(b1v - (float)b1h);
    }

    // bf16x3: Ah*Bl + Al*Bh + Ah*Bh  (small terms first) ~= fp32 product
    acc0 = __builtin_amdgcn_wmma_f32_16x16x32_bf16(false, Ah, false, B0l,
                                                   (short)0, acc0, false, false);
    acc0 = __builtin_amdgcn_wmma_f32_16x16x32_bf16(false, Al, false, B0h,
                                                   (short)0, acc0, false, false);
    acc0 = __builtin_amdgcn_wmma_f32_16x16x32_bf16(false, Ah, false, B0h,
                                                   (short)0, acc0, false, false);
    acc1 = __builtin_amdgcn_wmma_f32_16x16x32_bf16(false, Ah, false, B1l,
                                                   (short)0, acc1, false, false);
    acc1 = __builtin_amdgcn_wmma_f32_16x16x32_bf16(false, Al, false, B1h,
                                                   (short)0, acc1, false, false);
    acc1 = __builtin_amdgcn_wmma_f32_16x16x32_bf16(false, Ah, false, B1h,
                                                   (short)0, acc1, false, false);
  }

  // ---- Diagonal-band extraction via LDS, then coalesced row stores ----
  // C/D layout: element v, lane l -> (m = v + 8*(l>>4), n = l&15).
  // tile0: dx+4 = n - m in [0,8];  tile1: dx+4 = 16 + n - m in [0,8].
  // For each m, the two ranges are disjoint and cover 0..8 exactly once.
  __shared__ float tile[kP][16 + 1];  // +1 pad vs bank conflicts
#pragma unroll
  for (int v = 0; v < 8; ++v) {
    const int m  = v + 8 * half;
    const int d0 = lm - m;
    if (d0 >= 0 && d0 <= 8) tile[d0][m] = acc0[v];
    const int d1 = 16 + lm - m;
    if (d1 >= 0 && d1 <= 8) tile[d1][m] = acc1[v];
  }
  __syncthreads();

  // out[b, dyi*9 + dxi, y, x0 + m]
  float* obase = out + (((long)b * kPP + (long)dyi * kP) * kH + y) * kW + x0;
#pragma unroll
  for (int t = 0; t < 5; ++t) {
    const int idx = t * 32 + lane;  // 9*16 = 144 elements
    if (idx < kP * 16) {
      const int r = idx >> 4;
      const int m = idx & 15;
      obase[(long)r * kPlane + m] = tile[r][m];
    }
  }
}

extern "C" void kernel_launch(void* const* d_in, const int* in_sizes, int n_in,
                              void* d_out, int out_size, void* d_ws, size_t ws_size,
                              hipStream_t stream) {
  (void)in_sizes; (void)n_in; (void)out_size; (void)d_ws; (void)ws_size;
  const float* f0 = (const float*)d_in[0];
  const float* f1 = (const float*)d_in[1];
  float* out = (float*)d_out;

  const int nblk = kB * kH * kXT * kP;  // 4*96*10*9 = 34560 waves
  hipLaunchKernelGGL(corr_wmma_bf16x3, dim3(nblk), dim3(32), 0, stream,
                     f0, f1, out);
}
